// SparseCNN_6700148981987
// MI455X (gfx1250) — matrix operand order repaired and checked by reading, not compiled, not run on hardware
//
#include <hip/hip_runtime.h>
#include <hip/hip_fp16.h>

typedef __attribute__((ext_vector_type(16))) _Float16 v16h;
typedef __attribute__((ext_vector_type(8)))  float    v8f;
typedef __attribute__((ext_vector_type(4)))  unsigned int v4u;

// ---------------------------------------------------------------- utilities

__global__ void k_zero_u32(unsigned int* __restrict__ p, long n) {
    long i = blockIdx.x * (long)blockDim.x + threadIdx.x;
    long T = (long)gridDim.x * blockDim.x;
    for (; i < n; i += T) p[i] = 0u;
}

// ------------------------------------------------------- voxel scatter + mask

__global__ void k_scatter(const float* __restrict__ feats,
                          const int* __restrict__ coors,
                          _Float16* __restrict__ buf, float* __restrict__ mask,
                          int N, int D, int H, int W, int C) {
    int i = blockIdx.x * blockDim.x + threadIdx.x;
    if (i >= N) return;
    int b = coors[i * 4 + 0], z = coors[i * 4 + 1];
    int y = coors[i * 4 + 2], x = coors[i * 4 + 3];
    long pos = (((long)b * D + z) * H + y) * (long)W + x;
    mask[pos] = 1.0f;
    __half2* dst = (__half2*)(buf + pos * C);
    for (int c = 0; c < C; c += 2) {
        __half2 v = __floats2half2_rn(feats[(long)i * C + c], feats[(long)i * C + c + 1]);
        atomicAdd(dst + (c >> 1), v);
    }
}

// -------------------------------------------------- weight pack: fp32 -> f16
// packed layout: [chunk q][n (global out-ch)][32 k-values], zero padded.
// k_global = q*32 + t ; tap = k>>ciShift ; c = k&(Ci-1) ; w is (kd,kh,kw,Ci,Co).

__global__ void k_pack_w(const float* __restrict__ w, _Float16* __restrict__ wp,
                         int ciShift, int Co, int taps, int numChunks) {
    int idx = blockIdx.x * blockDim.x + threadIdx.x;
    int total = numChunks * Co * 32;
    if (idx >= total) return;
    int Ci = 1 << ciShift;
    int t = idx & 31;
    int rest = idx >> 5;
    int n = rest % Co;
    int q = rest / Co;
    int kg = q * 32 + t;
    float v = 0.0f;
    if (kg < taps * Ci) {
        int tap = kg >> ciShift, c = kg & (Ci - 1);
        v = w[((long)tap * Ci + c) * Co + n];
    }
    wp[idx] = (_Float16)v;
}

// ------------------------------------------------ implicit-GEMM conv (WMMA)
// One wave => 16 spatial (along W_out) x 16 out-channel tile, f32 accum.
// grid: x = x-n tiles / 8 waves, y = Ho, z = B*Do. Compile-time KH,KW,Ci.

template <int KH, int KW, int CISHIFT>
__global__ void __launch_bounds__(256)
k_conv_wmma(const _Float16* __restrict__ in, _Float16* __restrict__ out,
            const _Float16* __restrict__ wp,
            int Di, int Hi, int Wi,
            int Do, int Ho, int Wo, int Co,
            int sd, int sh, int sw, int pd, int ph, int pw,
            int numChunks, int Ktotal, int tilesX, int ntShift) {
    constexpr int Ci  = 1 << CISHIFT;
    constexpr int KHW = KH * KW;

    int wib  = threadIdx.x >> 5;                 // wave in block (0..7)
    int tile = blockIdx.x * 8 + wib;
    int nt = tile & ((1 << ntShift) - 1);
    int xt = tile >> ntShift;
    if (xt >= tilesX) return;                    // wave-uniform

    int lane = threadIdx.x & 31;
    int lo = lane & 15, hi = lane >> 4;

    int yo = blockIdx.y;
    int zo = blockIdx.z % Do;
    int b  = blockIdx.z / Do;
    int x0 = xt * 16;

    v8f acc = {};
    for (int q = 0; q < numChunks; ++q) {
        union { v4u u[2]; v16h v; } A;
        union { v4u u[2]; v16h v; } Bm;
#pragma unroll
        for (int r = 0; r < 2; ++r) {
            int k0 = q * 32 + 16 * r + 8 * hi;   // first K of this 8-half run
            v4u val = (v4u)0;
            if (k0 < Ktotal) {
                int tap = k0 >> CISHIFT;
                int c0  = k0 & (Ci - 1);
                int dz  = tap / KHW;             // compile-time divisor
                int rem = tap % KHW;
                int dy  = rem / KW, dx = rem % KW;
                int zi = zo * sd + dz - pd;
                int yi = yo * sh + dy - ph;
                int xi = (x0 + lo) * sw + dx - pw;
                if (zi >= 0 && zi < Di && yi >= 0 && yi < Hi && xi >= 0 && xi < Wi) {
                    const _Float16* p = in +
                        ((((long)b * Di + zi) * Hi + yi) * (long)Wi + xi) * Ci + c0;
                    val = *(const v4u*)p;        // 8 contiguous halves (16B)
                }
            }
            A.u[r] = val;
        }
        // B fragment: col n = lo, K-runs 8*hi..8*hi+7 and 16+8*hi..16+8*hi+7
        const _Float16* bp = wp + ((long)q * Co + (nt * 16 + lo)) * 32 + 8 * hi;
        Bm.u[0] = *(const v4u*)bp;
        Bm.u[1] = *(const v4u*)(bp + 16);

        acc = __builtin_amdgcn_wmma_f32_16x16x32_f16(
            false, A.v, false, Bm.v, (short)0, acc, false, false);
    }

    int n = nt * 16 + lo;
#pragma unroll
    for (int r = 0; r < 8; ++r) {
        int m = r + 8 * hi;
        int x = x0 + m;
        if (x < Wo)
            out[((((long)b * Do + zo) * Ho + yo) * (long)Wo + x) * Co + n] =
                (_Float16)acc[r];
    }
}

// ------------------------------------------------------ mask downsample (any)

__global__ void k_mask_down(const float* __restrict__ mi, float* __restrict__ mo,
                            int B, int Di, int Hi, int Wi,
                            int Do, int Ho, int Wo,
                            int kd, int kh, int kw,
                            int sd, int sh, int sw, int pd, int ph, int pw) {
    long total = (long)B * Do * Ho * Wo;
    long i = blockIdx.x * (long)blockDim.x + threadIdx.x;
    long T = (long)gridDim.x * blockDim.x;
    for (; i < total; i += T) {
        long t = i;
        int x = t % Wo; t /= Wo;
        int y = t % Ho; t /= Ho;
        int z = t % Do; t /= Do;
        int b = (int)t;
        float any = 0.0f;
        for (int dz = 0; dz < kd; ++dz) {
            int zi = z * sd + dz - pd;
            if (zi < 0 || zi >= Di) continue;
            for (int dy = 0; dy < kh; ++dy) {
                int yi = y * sh + dy - ph;
                if (yi < 0 || yi >= Hi) continue;
                for (int dx = 0; dx < kw; ++dx) {
                    int xi = x * sw + dx - pw;
                    if (xi < 0 || xi >= Wi) continue;
                    if (mi[(((long)b * Di + zi) * Hi + yi) * (long)Wi + xi] != 0.0f)
                        any = 1.0f;
                }
            }
        }
        mo[i] = any;
    }
}

// ----------------------------------------- masked BN: reduce / finalize / apply
// stats layout (floats): [0..63] sum, [64..127] sumsq, [128] n, [129..192] scale,
// [193..256] shift.

__global__ void k_bn_reduce(const _Float16* __restrict__ h,
                            const float* __restrict__ mask,
                            float* __restrict__ stats, long P, int cShift) {
    int C = 1 << cShift;
    long tid = blockIdx.x * (long)blockDim.x + threadIdx.x;
    long T = (long)gridDim.x * blockDim.x;       // multiple of C
    int c = (int)(tid & (C - 1));
    float s = 0.0f, s2 = 0.0f, cnt = 0.0f;
    for (long pos = tid >> cShift; pos < P; pos += (T >> cShift)) {
        float m = mask[pos];
        if (m != 0.0f) {
            float x = (float)h[(pos << cShift) + c];
            s += x; s2 += x * x;
            cnt += 1.0f;
        }
    }
    atomicAdd(&stats[c], s);
    atomicAdd(&stats[64 + c], s2);
    if (c == 0) atomicAdd(&stats[128], cnt);
}

__global__ void k_bn_finalize(float* __restrict__ stats,
                              const float* __restrict__ gamma,
                              const float* __restrict__ beta,
                              int C, float eps) {
    int c = threadIdx.x;
    if (c >= C) return;
    float n = stats[128];
    if (n < 1.0f) n = 1.0f;
    float mean = stats[c] / n;
    float var  = stats[64 + c] / n - mean * mean;
    float a = gamma[c] * rsqrtf(var + eps);
    stats[129 + c] = a;
    stats[193 + c] = beta[c] - mean * a;
}

__global__ void k_bn_apply(_Float16* __restrict__ h, const float* __restrict__ mask,
                           const float* __restrict__ stats, long P, int cShift) {
    int C = 1 << cShift;
    long total = P << cShift;
    long i = blockIdx.x * (long)blockDim.x + threadIdx.x;
    long T = (long)gridDim.x * blockDim.x;
    for (; i < total; i += T) {
        long pos = i >> cShift;
        int c = (int)(i & (C - 1));
        float x = (float)h[i];
        float y = fmaxf(stats[129 + c] * x + stats[193 + c], 0.0f) * mask[pos];
        h[i] = (_Float16)y;
    }
}

// --------------------------------------------------- final transpose to fp32
// out[b][c*Do+d][y][x] = h[b][d][y][x][c]

__global__ void k_final(const _Float16* __restrict__ h, float* __restrict__ out,
                        int B, int Do, int Ho, int Wo, int Co) {
    long total = (long)B * Co * Do * Ho * Wo;
    long i = blockIdx.x * (long)blockDim.x + threadIdx.x;
    long T = (long)gridDim.x * blockDim.x;
    for (; i < total; i += T) {
        long t = i;
        int x = t % Wo; t /= Wo;
        int y = t % Ho; t /= Ho;
        int d = t % Do; t /= Do;
        int c = t % Co; t /= Co;
        int b = (int)t;
        out[i] = (float)h[((((long)b * Do + d) * Ho + y) * (long)Wo + x) * Co + c];
    }
}

// ----------------------------------------------------------------- host side

struct LayerDef { int subm, ci, co, k0, k1, k2, s0, s1, s2, p0, p1, p2; };
static const LayerDef LDEF[14] = {
    {1, 16, 16, 3, 3, 3, 1, 1, 1, 1, 1, 1},
    {1, 16, 16, 3, 3, 3, 1, 1, 1, 1, 1, 1},
    {0, 16, 32, 3, 3, 3, 2, 2, 2, 1, 1, 1},
    {1, 32, 32, 3, 3, 3, 1, 1, 1, 1, 1, 1},
    {1, 32, 32, 3, 3, 3, 1, 1, 1, 1, 1, 1},
    {0, 32, 64, 3, 3, 3, 2, 2, 2, 1, 1, 1},
    {1, 64, 64, 3, 3, 3, 1, 1, 1, 1, 1, 1},
    {1, 64, 64, 3, 3, 3, 1, 1, 1, 1, 1, 1},
    {1, 64, 64, 3, 3, 3, 1, 1, 1, 1, 1, 1},
    {0, 64, 64, 3, 3, 3, 2, 2, 2, 0, 1, 1},
    {1, 64, 64, 3, 3, 3, 1, 1, 1, 1, 1, 1},
    {1, 64, 64, 3, 3, 3, 1, 1, 1, 1, 1, 1},
    {1, 64, 64, 3, 3, 3, 1, 1, 1, 1, 1, 1},
    {0, 64, 64, 3, 1, 1, 2, 1, 1, 0, 0, 0},
};

static inline int outDim(int in, int k, int s, int p) { return (in + 2 * p - k) / s + 1; }
static inline int ilog2(int v) { int s = 0; while ((1 << s) < v) ++s; return s; }
static inline int gblocks(long n, int tb) {
    long g = (n + tb - 1) / tb;
    if (g > 65535L * 16L) g = 65535L * 16L;
    return (int)g;
}

extern "C" void kernel_launch(void* const* d_in, const int* in_sizes, int n_in,
                              void* d_out, int out_size, void* d_ws, size_t ws_size,
                              hipStream_t stream) {
    (void)in_sizes; (void)n_in; (void)out_size; (void)ws_size;
    const float* feats = (const float*)d_in[0];
    const int*   coors = (const int*)d_in[1];
    // params flattened: w[0..13] at 2.., gamma[0..13] at 16.., beta[0..13] at 30..
    const int N_VOX = 80000;
    const float EPS = 1e-3f;

    char* ws = (char*)d_ws;
    _Float16* bufA  = (_Float16*)(ws);
    _Float16* bufB  = (_Float16*)(ws + (size_t)96 * 1024 * 1024);
    float*    maskA = (float*)   (ws + (size_t)192 * 1024 * 1024);
    float*    maskB = (float*)   (ws + (size_t)204 * 1024 * 1024);
    _Float16* wpack = (_Float16*)(ws + (size_t)216 * 1024 * 1024);
    float*    stats = (float*)   (ws + (size_t)217 * 1024 * 1024);

    int B = 2, D = 41, H = 200, W = 176, C = 16;
    long P = (long)B * D * H * W;

    // zero input activation buffer + mask, then scatter voxels
    k_zero_u32<<<gblocks(P * C / 2, 256), 256, 0, stream>>>((unsigned int*)bufA, P * C / 2);
    k_zero_u32<<<gblocks(P, 256), 256, 0, stream>>>((unsigned int*)maskA, P);
    k_scatter<<<(N_VOX + 255) / 256, 256, 0, stream>>>(feats, coors, bufA, maskA,
                                                       N_VOX, D, H, W, C);

    _Float16* cur = bufA, *nxt = bufB;
    float* mcur = maskA, *mnxt = maskB;

    for (int L = 0; L < 14; ++L) {
        const LayerDef& ld = LDEF[L];
        const float* w     = (const float*)d_in[2 + L];
        const float* gamma = (const float*)d_in[16 + L];
        const float* beta  = (const float*)d_in[30 + L];

        int Do_ = ld.subm ? D : outDim(D, ld.k0, ld.s0, ld.p0);
        int Ho_ = ld.subm ? H : outDim(H, ld.k1, ld.s1, ld.p1);
        int Wo_ = ld.subm ? W : outDim(W, ld.k2, ld.s2, ld.p2);
        long Po = (long)B * Do_ * Ho_ * Wo_;

        int taps = ld.k0 * ld.k1 * ld.k2;
        int Kt = taps * ld.ci;
        int nChunks = (Kt + 31) / 32;
        int ciShift = ilog2(ld.ci);
        int coShift = ilog2(ld.co);

        // pack weights
        int packTot = nChunks * ld.co * 32;
        k_pack_w<<<(packTot + 255) / 256, 256, 0, stream>>>(w, wpack, ciShift, ld.co,
                                                            taps, nChunks);

        // conv via WMMA
        int tilesX = (Wo_ + 15) / 16;
        int nTiles = ld.co / 16;
        int ntShift = ilog2(nTiles);
        dim3 cgrid((tilesX * nTiles + 7) / 8, Ho_, B * Do_);
        if (ld.k1 == 3) {
            if (ld.ci == 16)
                k_conv_wmma<3, 3, 4><<<cgrid, 256, 0, stream>>>(
                    cur, nxt, wpack, D, H, W, Do_, Ho_, Wo_, ld.co,
                    ld.s0, ld.s1, ld.s2, ld.p0, ld.p1, ld.p2,
                    nChunks, Kt, tilesX, ntShift);
            else if (ld.ci == 32)
                k_conv_wmma<3, 3, 5><<<cgrid, 256, 0, stream>>>(
                    cur, nxt, wpack, D, H, W, Do_, Ho_, Wo_, ld.co,
                    ld.s0, ld.s1, ld.s2, ld.p0, ld.p1, ld.p2,
                    nChunks, Kt, tilesX, ntShift);
            else
                k_conv_wmma<3, 3, 6><<<cgrid, 256, 0, stream>>>(
                    cur, nxt, wpack, D, H, W, Do_, Ho_, Wo_, ld.co,
                    ld.s0, ld.s1, ld.s2, ld.p0, ld.p1, ld.p2,
                    nChunks, Kt, tilesX, ntShift);
        } else {
            k_conv_wmma<1, 1, 6><<<cgrid, 256, 0, stream>>>(
                cur, nxt, wpack, D, H, W, Do_, Ho_, Wo_, ld.co,
                ld.s0, ld.s1, ld.s2, ld.p0, ld.p1, ld.p2,
                nChunks, Kt, tilesX, ntShift);
        }

        // mask update for strided conv layers
        float* muse = mcur;
        if (!ld.subm) {
            k_mask_down<<<gblocks(Po, 256), 256, 0, stream>>>(
                mcur, mnxt, B, D, H, W, Do_, Ho_, Wo_,
                ld.k0, ld.k1, ld.k2, ld.s0, ld.s1, ld.s2, ld.p0, ld.p1, ld.p2);
            muse = mnxt;
        }

        // masked BN + ReLU
        k_zero_u32<<<1, 256, 0, stream>>>((unsigned int*)stats, 129);
        k_bn_reduce<<<1024, 256, 0, stream>>>(nxt, muse, stats, Po, coShift);
        k_bn_finalize<<<1, 64, 0, stream>>>(stats, gamma, beta, ld.co, EPS);
        k_bn_apply<<<gblocks(Po << coShift, 256), 256, 0, stream>>>(nxt, muse, stats,
                                                                    Po, coShift);

        // ping-pong
        { _Float16* tmp = cur; cur = nxt; nxt = tmp; }
        if (!ld.subm) { float* tmp = mcur; mcur = mnxt; mnxt = tmp; }
        D = Do_; H = Ho_; W = Wo_; C = ld.co; P = Po;
    }

    // final: (B,D,H,W,C) f16 -> (B, C*D, H, W) fp32
    k_final<<<gblocks((long)B * C * D * H * W, 256), 256, 0, stream>>>(
        cur, (float*)d_out, B, D, H, W, C);
}
